// Hypergraph_63462436765972
// MI455X (gfx1250) — matrix operand order repaired and checked by reading, not compiled, not run on hardware
//
#include <hip/hip_runtime.h>
#include <cmath>

// ---------------------------------------------------------------------------
// CDNA5 (gfx1250, wave32) implementation of the hypergraph conv forward pass.
// Dense layers use v_wmma_f32_16x16x32_bf16 (fp32 accumulate).
// ---------------------------------------------------------------------------

typedef __attribute__((ext_vector_type(16))) __bf16 v16bf;
typedef __attribute__((ext_vector_type(8)))  __bf16 v8bf;
typedef __attribute__((ext_vector_type(8)))  float  v8f;

static constexpr int   kNV = 100000, kNE = 20000, kE = 800000;
static constexpr int   kDIN = 128, kH = 128, kNCLS = 40, kNLAYERS = 2;
static constexpr int   kCH = 65536;          // edge-chunk size
static constexpr float kAlpha = 0.4f;
static constexpr float kEps = 1e-5f;

__device__ __forceinline__ __bf16 f2bf(float f) {
  unsigned u = __builtin_bit_cast(unsigned, f);
  unsigned r = u + 0x7FFFu + ((u >> 16) & 1u);   // round-to-nearest-even
  unsigned short h = (unsigned short)(r >> 16);
  return __builtin_bit_cast(__bf16, h);
}

// ------------------------------ elementwise --------------------------------

__global__ void cvt_bf16_kernel(const float* __restrict__ src,
                                __bf16* __restrict__ dst, long n) {
  long i = (long)blockIdx.x * blockDim.x + threadIdx.x;
  if (i < n) dst[i] = f2bf(src[i]);
}

__global__ void relu_kernel(const float* __restrict__ src,
                            float* __restrict__ dst, long n) {
  long i = (long)blockIdx.x * blockDim.x + threadIdx.x;
  if (i < n) dst[i] = fmaxf(src[i], 0.0f);
}

// out = (1-alpha)*acc/max(cnt,1) + alpha*base
__global__ void finish_mean_blend_kernel(const float* __restrict__ acc,
                                         const float* __restrict__ cnt,
                                         const float* __restrict__ base,
                                         float alpha, float* __restrict__ out,
                                         long rows, int H) {
  long i = (long)blockIdx.x * blockDim.x + threadIdx.x;
  long total = rows * H;
  if (i >= total) return;
  long r = i / H;
  float c = fmaxf(cnt[r], 1.0f);
  out[i] = (1.0f - alpha) * acc[i] / c + alpha * base[i];
}

// t = beta*t + (1-beta)*ehold[eidx[j0+r]]
__global__ void blend_gather_kernel(float* __restrict__ t,
                                    const float* __restrict__ ehold,
                                    const int* __restrict__ eidx, int j0,
                                    float beta, long total, int H) {
  long i = (long)blockIdx.x * blockDim.x + threadIdx.x;
  if (i >= total) return;
  long r = i / H;
  int  h = (int)(i - r * H);
  int  e = eidx[j0 + r];
  t[i] = beta * t[i] + (1.0f - beta) * ehold[(long)e * H + h];
}

// out = relu(beta*a + (1-beta)*b)
__global__ void blend_relu_kernel(const float* __restrict__ a,
                                  const float* __restrict__ b, float beta,
                                  float* __restrict__ out, long n) {
  long i = (long)blockIdx.x * blockDim.x + threadIdx.x;
  if (i < n) out[i] = fmaxf(beta * a[i] + (1.0f - beta) * b[i], 0.0f);
}

// ------------------------- scatter (atomic mean) ---------------------------

// acc[eidx[j]] += nm[vidx[j]] ; cnt[eidx[j]] += 1      (one block per incidence)
__global__ void scatter_nm_kernel(const float* __restrict__ nm,
                                  const int* __restrict__ vidx,
                                  const int* __restrict__ eidx,
                                  float* __restrict__ acc,
                                  float* __restrict__ cnt, int E, int H) {
  int j = blockIdx.x;
  if (j >= E) return;
  int h = threadIdx.x;
  int sv = vidx[j], de = eidx[j];
  atomicAdd(&acc[(long)de * H + h], nm[(long)sv * H + h]);
  if (h == 0) atomicAdd(&cnt[de], 1.0f);
}

// acc[idx[j0+r]] += src[r] ; cnt += 1
__global__ void scatter_add_chunk_kernel(const float* __restrict__ src,
                                         const int* __restrict__ idx, int j0,
                                         float* __restrict__ acc,
                                         float* __restrict__ cnt, int cn, int H) {
  int r = blockIdx.x;
  if (r >= cn) return;
  int h = threadIdx.x;
  int d = idx[j0 + r];
  atomicAdd(&acc[(long)d * H + h], src[(long)r * H + h]);
  if (h == 0) atomicAdd(&cnt[d], 1.0f);
}

// --------------------------- LayerNorm (wave/row) --------------------------

// Y = [relu?] LN(X*rowscale; g,b) -> bf16, row stride D. One wave32 per row.
__global__ void __launch_bounds__(256)
ln_bf16_kernel(const float* __restrict__ X, const float* __restrict__ rowscale,
               const float* __restrict__ g, const float* __restrict__ b,
               __bf16* __restrict__ Y, int M, int D, int relu) {
  int wave = threadIdx.x >> 5, lane = threadIdx.x & 31;
  long row = (long)blockIdx.x * 8 + wave;
  if (row >= M) return;
  const float* x = X + row * D;
  float s = rowscale ? rowscale[row] : 1.0f;
  float sum = 0.0f, sq = 0.0f;
  for (int i = lane; i < D; i += 32) {
    float v = x[i] * s;
    sum += v; sq += v * v;
  }
  for (int off = 16; off; off >>= 1) {
    sum += __shfl_xor(sum, off, 32);
    sq  += __shfl_xor(sq,  off, 32);
  }
  float mean = sum / D;
  float var  = sq / D - mean * mean;
  float rs   = rsqrtf(var + kEps);
  for (int i = lane; i < D; i += 32) {
    float v = (x[i] * s - mean) * rs * g[i] + b[i];
    if (relu) v = fmaxf(v, 0.0f);
    Y[row * D + i] = f2bf(v);
  }
}

// cat = [vh[vidx[j]], edge[eidx[j]]]; out = LN(cat; g0,b0) -> bf16 [cn,256]
__global__ void __launch_bounds__(256)
gather_concat_ln_kernel(const float* __restrict__ vh,
                        const float* __restrict__ edge,
                        const int* __restrict__ vidx,
                        const int* __restrict__ eidx,
                        const float* __restrict__ g0,
                        const float* __restrict__ b0,
                        __bf16* __restrict__ out, int j0, int cn) {
  int wave = threadIdx.x >> 5, lane = threadIdx.x & 31;
  int r = blockIdx.x * 8 + wave;
  if (r >= cn) return;
  int j = j0 + r;
  const float* xa = vh   + (long)vidx[j] * kH;
  const float* xb = edge + (long)eidx[j] * kH;
  float vals[8];
  float sum = 0.0f, sq = 0.0f;
#pragma unroll
  for (int t = 0; t < 8; ++t) {
    int i = lane + t * 32;                    // 0..255
    float v = (i < kH) ? xa[i] : xb[i - kH];
    vals[t] = v;
    sum += v; sq += v * v;
  }
  for (int off = 16; off; off >>= 1) {
    sum += __shfl_xor(sum, off, 32);
    sq  += __shfl_xor(sq,  off, 32);
  }
  float mean = sum * (1.0f / 256.0f);
  float var  = sq  * (1.0f / 256.0f) - mean * mean;
  float rs   = rsqrtf(var + kEps);
#pragma unroll
  for (int t = 0; t < 8; ++t) {
    int i = lane + t * 32;
    out[(long)r * 256 + i] = f2bf((vals[t] - mean) * rs * g0[i] + b0[i]);
  }
}

// ------------------------------ WMMA GEMM ----------------------------------
// C[M,N] = A[M,K](bf16,lda) @ B[K,N](bf16,ldb) + bias, optional relu.
// Block = 8 wave32s. Block tile = 32 rows x 128 cols; wave w owns cols
// [16w,16w+16) and TWO 16x16 M-tiles sharing one B fragment.
// A tile (32x32) row-major in LDS; B tile stored TRANSPOSED (n-major) so each
// fragment is two contiguous 16B runs -> ds_load_b128. Rows padded to 40
// elements (80B = 20 banks) for conflict-free 128b lane reads.
// Requires: N <= 128, K % 32 == 0, 16B-aligned A/B with even lda/ldb.
#define TILE_K 32

union Frag16 { v16bf v; v8bf h[2]; };

__global__ void __launch_bounds__(256)
gemm_bf16_wmma_kernel(const __bf16* __restrict__ A, int lda,
                      const __bf16* __restrict__ B, int ldb,
                      const float* __restrict__ bias,
                      float* __restrict__ C, int ldc,
                      int M, int N, int K, int relu) {
  __shared__ __bf16 As[32][40];      // [m][k], rows 80B apart
  __shared__ __bf16 Bst[128][40];    // [n][k] (transposed tile)

  int tid  = threadIdx.x;
  int wave = tid >> 5, lane = tid & 31;
  int half = lane >> 4, l16 = lane & 15;
  int m0 = blockIdx.x * 32;
  int n0 = wave * 16;
  int nch = (N + 7) >> 3;            // 8-col chunks per B k-row

  v8f acc0 = {}, acc1 = {};

  for (int k0 = 0; k0 < K; k0 += TILE_K) {
    // ---- stage A tile 32x32: 128 chunks of 8 bf16 (16B vector loads) ----
    if (tid < 128) {
      int r = tid >> 2, c = (tid & 3) * 8;
      int gm = m0 + r;
      v8bf val = {};
      if (gm < M) val = *(const v8bf*)(A + (long)gm * lda + k0 + c);
      *(v8bf*)&As[r][c] = val;
    }
    // ---- stage B tile 32xN transposed into Bst[n][k] ----
    for (int idx = tid; idx < TILE_K * nch; idx += 256) {
      int r = idx / nch;                   // k-row within tile
      int c = (idx - r * nch) * 8;         // first col of chunk
      const __bf16* src = B + (long)(k0 + r) * ldb + c;
      __bf16 tmp[8];
      if (c + 8 <= N) {
        *(v8bf*)tmp = *(const v8bf*)src;
      } else {
#pragma unroll
        for (int j = 0; j < 8; ++j) tmp[j] = (c + j < N) ? src[j] : (__bf16)0.0f;
      }
#pragma unroll
      for (int j = 0; j < 8; ++j) Bst[c + j][r] = tmp[j];
    }
    if (k0 + TILE_K < K)  // pull next A strip toward the caches
      __builtin_prefetch(A + (long)(m0 + (tid & 31)) * lda + k0 + TILE_K, 0, 1);
    __syncthreads();

    // ---- fragments (ISA 7.12.2 16-bit layouts; contiguous runs) ----
    // A lane(l16,half): row l16 (+16 for tile1), K runs [8h,8h+8) and [16+8h,+8)
    // B lane(l16,half): n = n0+l16, K runs [8h,8h+8) and [16+8h,+8)
    Frag16 a0, a1, bf;
    a0.h[0] = *(const v8bf*)&As[l16][half * 8];
    a0.h[1] = *(const v8bf*)&As[l16][16 + half * 8];
    a1.h[0] = *(const v8bf*)&As[16 + l16][half * 8];
    a1.h[1] = *(const v8bf*)&As[16 + l16][16 + half * 8];
    bf.h[0] = *(const v8bf*)&Bst[n0 + l16][half * 8];
    bf.h[1] = *(const v8bf*)&Bst[n0 + l16][16 + half * 8];

    acc0 = __builtin_amdgcn_wmma_f32_16x16x32_bf16(false, a0.v, false, bf.v,
                                                   (short)0, acc0, false, false);
    acc1 = __builtin_amdgcn_wmma_f32_16x16x32_bf16(false, a1.v, false, bf.v,
                                                   (short)0, acc1, false, false);
    __syncthreads();
  }

  // ---- epilogue: C layout VGPR r -> M = r + 8*half, N = l16 ----
  int col = n0 + l16;
  if (col < N) {
    float bv = bias ? bias[col] : 0.0f;
#pragma unroll
    for (int r = 0; r < 8; ++r) {
      int row = m0 + r + half * 8;
      if (row < M) {
        float v = acc0[r] + bv;
        if (relu) v = fmaxf(v, 0.0f);
        C[(long)row * ldc + col] = v;
      }
      int row1 = m0 + 16 + r + half * 8;
      if (row1 < M) {
        float v = acc1[r] + bv;
        if (relu) v = fmaxf(v, 0.0f);
        C[(long)row1 * ldc + col] = v;
      }
    }
  }
}

// ------------------------------- host side ---------------------------------

struct MlpP { const float *W1, *c1, *g1, *b1, *W2, *c2, *g0, *b0; };

extern "C" void kernel_launch(void* const* d_in, const int* in_sizes, int n_in,
                              void* d_out, int out_size, void* d_ws, size_t ws_size,
                              hipStream_t stream) {
  (void)in_sizes; (void)n_in; (void)out_size; (void)ws_size;

  // ---- inputs, in setup_inputs() insertion order (params flattened in dict
  // insertion order; _mlp_params appends g0/b0 after c2) ----
  const float* v        = (const float*)d_in[0];
  const int*   vidx     = (const int*)  d_in[2];
  const int*   eidx     = (const int*)  d_in[3];
  const float* n_weight = (const float*)d_in[4];
  const float* node_map_W = (const float*)d_in[5];
  const float* node_map_b = (const float*)d_in[6];
  const float* edge_emb   = (const float*)d_in[7];

  auto mlpAt = [&](int base, bool norm) {
    MlpP p;
    p.W1 = (const float*)d_in[base + 0]; p.c1 = (const float*)d_in[base + 1];
    p.g1 = (const float*)d_in[base + 2]; p.b1 = (const float*)d_in[base + 3];
    p.W2 = (const float*)d_in[base + 4]; p.c2 = (const float*)d_in[base + 5];
    p.g0 = norm ? (const float*)d_in[base + 6] : nullptr;
    p.b0 = norm ? (const float*)d_in[base + 7] : nullptr;
    return p;
  };
  MlpP cls = mlpAt(8 + kNLAYERS * 24, false);

  // ---- workspace carve-up (~450 MB total) ----
  float* f = (float*)d_ws;
  float* v0   = f; f += (long)kNV * kH;
  float* e0   = f; f += (long)kNE * kH;
  float* vh   = f; f += (long)kNV * kH;
  float* eh   = f; f += (long)kNE * kH;
  float* T1   = f; f += (long)kNV * kH;   // node scratch / "node" pre-blend
  float* NM   = f; f += (long)kNV * kH;   // node_msg / node_agg output
  float* EDG  = f; f += (long)kNE * kH;   // new edge (pre-relu)
  float* eacc = f; f += (long)kNE * kH;
  float* ecnt = f; f += kNE;
  float* vacc = f; f += (long)kNV * kH;
  float* vcnt = f; f += kNV;
  float* T1c  = f; f += (long)kCH * kH;   // chunk hidden (f32)
  float* T2c  = f; f += (long)kCH * kH;   // chunk output (f32)
  __bf16* XB   = (__bf16*)f;              // LN outputs (bf16), NV*H elems
  __bf16* CATB = XB + (long)kNV * kH;     // concat chunk, CH*2H elems
  __bf16* VB   = CATB + (long)kCH * 2 * kH; // v in bf16
  __bf16* wpool = VB + (long)kNV * kDIN;  // bf16 weight pool

  auto cvt = [&](const float* src, __bf16* dst, long n) {
    cvt_bf16_kernel<<<dim3((unsigned)((n + 255) / 256)), dim3(256), 0, stream>>>(src, dst, n);
  };
  auto walloc = [&](const float* src, long n) {
    __bf16* p = wpool; wpool += n; cvt(src, p, n); return p;
  };

  // ---- weight conversions (tiny, once per launch) ----
  __bf16* nmWb = walloc(node_map_W, (long)kDIN * kH);
  __bf16 *n2mW1b[kNLAYERS], *n2mW2b[kNLAYERS], *e2mW1b[kNLAYERS],
         *e2mW2b[kNLAYERS], *nagW1b[kNLAYERS], *nagW2b[kNLAYERS];
  for (int i = 0; i < kNLAYERS; ++i) {
    int base = 8 + i * 24;
    MlpP n2m = mlpAt(base + 0, true);
    MlpP e2m = mlpAt(base + 8, true);
    MlpP nag = mlpAt(base + 16, true);
    n2mW1b[i] = walloc(n2m.W1, (long)kH * kH);
    n2mW2b[i] = walloc(n2m.W2, (long)kH * kH);
    e2mW1b[i] = walloc(e2m.W1, (long)2 * kH * kH);
    e2mW2b[i] = walloc(e2m.W2, (long)kH * kH);
    nagW1b[i] = walloc(nag.W1, (long)kH * kH);
    nagW2b[i] = walloc(nag.W2, (long)kH * kH);
  }
  __bf16* clsW1b = walloc(cls.W1, (long)kH * (kH / 2));
  __bf16* clsW2b = walloc(cls.W2, (long)(kH / 2) * kNCLS);

  auto gemm = [&](const __bf16* A, int lda, const __bf16* B, int ldb,
                  const float* bias, float* C, int ldc,
                  int M, int N, int K, int relu) {
    gemm_bf16_wmma_kernel<<<dim3((unsigned)((M + 31) / 32)), dim3(256), 0, stream>>>(
        A, lda, B, ldb, bias, C, ldc, M, N, K, relu);
  };
  auto ln = [&](const float* X, const float* rs, const float* g, const float* b,
                __bf16* Y, int M, int D, int relu) {
    ln_bf16_kernel<<<dim3((unsigned)((M + 7) / 8)), dim3(256), 0, stream>>>(
        X, rs, g, b, Y, M, D, relu);
  };
  auto ewgrid = [](long n) { return dim3((unsigned)((n + 255) / 256)); };

  // ---- v0 = relu(v @ node_map_W + b); e0 = relu(edge_emb) ----
  cvt(v, VB, (long)kNV * kDIN);
  gemm(VB, kDIN, nmWb, kH, node_map_b, v0, kH, kNV, kH, kDIN, 1);
  relu_kernel<<<ewgrid((long)kNE * kH), 256, 0, stream>>>(edge_emb, e0, (long)kNE * kH);
  (void)hipMemcpyAsync(vh, v0, (long)kNV * kH * sizeof(float), hipMemcpyDeviceToDevice, stream);
  (void)hipMemcpyAsync(eh, e0, (long)kNE * kH * sizeof(float), hipMemcpyDeviceToDevice, stream);

  // ---- conv layers ----
  for (int i = 0; i < kNLAYERS; ++i) {
    float beta = logf(0.5f / (float)(i + 1) + 1.0f);
    int base = 8 + i * 24;
    MlpP n2m = mlpAt(base + 0, true);
    MlpP e2m = mlpAt(base + 8, true);
    MlpP nag = mlpAt(base + 16, true);

    // node2msg: NM = MLP(vh * n_weight)  on all NV nodes
    ln(vh, n_weight, n2m.g0, n2m.b0, XB, kNV, kH, 0);
    gemm(XB, kH, n2mW1b[i], kH, n2m.c1, T1, kH, kNV, kH, kH, 0);
    ln(T1, nullptr, n2m.g1, n2m.b1, XB, kNV, kH, 1);
    gemm(XB, kH, n2mW2b[i], kH, n2m.c2, NM, kH, kNV, kH, kH, 0);

    // edge = (1-a)*scatter_mean(NM[vidx], eidx) + a*e0
    (void)hipMemsetAsync(eacc, 0, (long)kNE * kH * sizeof(float), stream);
    (void)hipMemsetAsync(ecnt, 0, (long)kNE * sizeof(float), stream);
    scatter_nm_kernel<<<dim3(kE), dim3(kH), 0, stream>>>(NM, vidx, eidx, eacc, ecnt, kE, kH);
    finish_mean_blend_kernel<<<ewgrid((long)kNE * kH), 256, 0, stream>>>(
        eacc, ecnt, e0, kAlpha, EDG, kNE, kH);

    // edge2msg over incidence chunks; scatter into node accumulators
    (void)hipMemsetAsync(vacc, 0, (long)kNV * kH * sizeof(float), stream);
    (void)hipMemsetAsync(vcnt, 0, (long)kNV * sizeof(float), stream);
    for (int j0 = 0; j0 < kE; j0 += kCH) {
      int cn = (kE - j0 < kCH) ? (kE - j0) : kCH;
      gather_concat_ln_kernel<<<dim3((unsigned)((cn + 7) / 8)), dim3(256), 0, stream>>>(
          vh, EDG, vidx, eidx, e2m.g0, e2m.b0, CATB, j0, cn);
      gemm(CATB, 2 * kH, e2mW1b[i], kH, e2m.c1, T1c, kH, cn, kH, 2 * kH, 0);
      ln(T1c, nullptr, e2m.g1, e2m.b1, XB, cn, kH, 1);
      gemm(XB, kH, e2mW2b[i], kH, e2m.c2, T2c, kH, cn, kH, kH, 0);
      blend_gather_kernel<<<ewgrid((long)cn * kH), 256, 0, stream>>>(
          T2c, eh, eidx, j0, beta, (long)cn * kH, kH);
      scatter_add_chunk_kernel<<<dim3(cn), dim3(kH), 0, stream>>>(
          T2c, vidx, j0, vacc, vcnt, cn, kH);
    }

    // node = (1-a)*mean + a*v0 ; node = beta*MLP(node) + (1-beta)*node
    finish_mean_blend_kernel<<<ewgrid((long)kNV * kH), 256, 0, stream>>>(
        vacc, vcnt, v0, kAlpha, T1, kNV, kH);
    ln(T1, nullptr, nag.g0, nag.b0, XB, kNV, kH, 0);
    gemm(XB, kH, nagW1b[i], kH, nag.c1, NM, kH, kNV, kH, kH, 0);
    ln(NM, nullptr, nag.g1, nag.b1, XB, kNV, kH, 1);
    gemm(XB, kH, nagW2b[i], kH, nag.c2, NM, kH, kNV, kH, kH, 0);

    blend_relu_kernel<<<ewgrid((long)kNV * kH), 256, 0, stream>>>(
        NM, T1, beta, vh, (long)kNV * kH);
    relu_kernel<<<ewgrid((long)kNE * kH), 256, 0, stream>>>(EDG, eh, (long)kNE * kH);
  }

  // ---- outputs: [vh | eh | pred] ----
  float* out = (float*)d_out;
  (void)hipMemcpyAsync(out, vh, (long)kNV * kH * sizeof(float), hipMemcpyDeviceToDevice, stream);
  (void)hipMemcpyAsync(out + (long)kNV * kH, eh, (long)kNE * kH * sizeof(float),
                       hipMemcpyDeviceToDevice, stream);

  // classifier: vh @ W1 + c1 -> relu(LN) -> @ W2 + c2
  cvt(vh, XB, (long)kNV * kH);
  gemm(XB, kH, clsW1b, kH / 2, cls.c1, T1, kH / 2, kNV, kH / 2, kH, 0);
  ln(T1, nullptr, cls.g1, cls.b1, XB, kNV, kH / 2, 1);
  gemm(XB, kH / 2, clsW2b, kNCLS, cls.c2, out + (long)kNV * kH + (long)kNE * kH,
       kNCLS, kNV, kNCLS, kH / 2, 0);
}
